// TwoLSTMPooler_24953759990486
// MI455X (gfx1250) — compile-verified, matrix operand order checked by reading, not run on hardware
//
#include <hip/hip_runtime.h>
#include <hip/hip_bf16.h>

typedef __bf16 bf16_t;
typedef __bf16 v16bf __attribute__((ext_vector_type(16)));
typedef float  v8f   __attribute__((ext_vector_type(8)));
typedef unsigned int v4u __attribute__((ext_vector_type(4)));
typedef int v4i __attribute__((ext_vector_type(4)));
typedef int v8i __attribute__((ext_vector_type(8)));

#define LL   13
#define BB   32
#define SS   512
#define HH   768
#define HLc  256
#define D1c  512
#define G4   1024          // 4*HL
#define NB   (BB * SS)     // 16384
#define TT2  511
#define M2   (TT2 * BB)    // 16352

__device__ __forceinline__ v8f wmma_bf16(v16bf a, v16bf b, v8f c) {
  // D = A(16x32 bf16) x B(32x16 bf16) + C(16x16 f32)
  return __builtin_amdgcn_wmma_f32_16x16x32_bf16(false, a, false, b, (short)0, c,
                                                 false, false);
}

__device__ __forceinline__ v16bf frag_ld(const bf16_t* p) {
  return *(const v16bf*)(const void*)p;
}

__device__ __forceinline__ float sigm(float x) { return 1.0f / (1.0f + __expf(-x)); }

// Issue `chunks` x 256-thread x 16B async global->LDS copies (ASYNCcnt path).
__device__ __forceinline__ void async_tile_copy(const void* g, void* lds, int tid,
                                                int chunks) {
  unsigned lbase = (unsigned)(size_t)lds;           // low 32 bits = LDS offset
  const char* gp = (const char*)g;
#pragma unroll
  for (int c = 0; c < 12; ++c) {
    if (c >= chunks) break;
    unsigned loff  = lbase + (unsigned)(tid * 16 + c * 4096);
    const char* ga = gp + tid * 16 + c * 4096;
    asm volatile("global_load_async_to_lds_b128 %0, %1, off"
                 :: "v"(loff), "v"(ga) : "memory");
  }
}
__device__ __forceinline__ void async_wait0() {
  asm volatile("s_wait_asynccnt 0" ::: "memory");
}

// TDM: load a 2D tile of bf16 (rows x rowlen, row-major, stride = rowlen) from
// global memory into LDS via the Tensor Data Mover (TENSORcnt path).
// D# layout per CDNA5 ISA.
// This toolchain's builtin takes 6 args: (v4u g0, v8i g1, v4i, v4i, v8i, i32 cpol).
__device__ __forceinline__ void tdm_load_2d_bf16(unsigned lds_off, const void* gptr,
                                                 unsigned rows, unsigned rowlen) {
  unsigned long long ga = (unsigned long long)(size_t)gptr;
  v4u g0;
  g0[0] = 1u;                                          // count=1, user descriptor
  g0[1] = lds_off;                                     // lds_addr
  g0[2] = (unsigned)(ga & 0xffffffffu);                // global_addr[31:0]
  g0[3] = (unsigned)((ga >> 32) & 0x01ffffffu)         // global_addr[56:32]
          | 0x80000000u;                               // type=2 ("image")
  v8i g1;
  g1[0] = (int)(1u << 16);                             // data_size=1 (2B), mask=0
  g1[1] = (int)((rowlen & 0xffffu) << 16);             // tensor_dim0[15:0]
  g1[2] = (int)(((rowlen >> 16) & 0xffffu)             // tensor_dim0[31:16]
                | ((rows & 0xffffu) << 16));           // tensor_dim1[15:0]
  g1[3] = (int)(((rows >> 16) & 0xffffu)               // tensor_dim1[31:16]
                | ((rowlen & 0xffffu) << 16));         // tile_dim0
  g1[4] = (int)(rows & 0xffffu);                       // tile_dim1 (tile_dim2=0)
  g1[5] = (int)rowlen;                                 // tensor_dim0_stride[31:0]
  g1[6] = 0;                                           // stride hi / dim1_stride
  g1[7] = 0;
  v4i gz4 = {0, 0, 0, 0};
  v8i gz8 = {0, 0, 0, 0, 0, 0, 0, 0};
  __builtin_amdgcn_tensor_load_to_lds(g0, g1, gz4, gz4, gz8, 0);
}
__device__ __forceinline__ void tensor_wait0() {
  __builtin_amdgcn_s_wait_tensorcnt(0);
}

// ---------------------------------------------------------------- f32 -> bf16
__global__ void k_cvt(const float* __restrict__ src, bf16_t* __restrict__ dst, int n) {
  int i = blockIdx.x * blockDim.x + threadIdx.x;
  if (i < n) dst[i] = (bf16_t)src[i];
}

// ---------------------------------------------------------------- LSTM1 (fused scan)
// grid (NB/32, 2 dirs), block 256 (8 waves). Block owns 32 batch rows (2 WMMA
// row tiles -> each B fragment feeds 2 WMMAs, halving L2 weight traffic) and
// runs the whole 13-step scan. Wave w owns cell columns [w*32, w*32+32): for
// each gate g it accumulates output columns g*256 + w*32 .. +32, so i/f/g/o
// for a cell column live in one wave and c never leaves accumulator registers.
// x tiles are double-buffered with async global->LDS DMA.
__global__ void __launch_bounds__(256) __attribute__((amdgpu_waves_per_eu(2)))
k_lstm1(const bf16_t* __restrict__ x,     // [13][16384][768] bf16
        const bf16_t* __restrict__ Wih,   // [2][1024][768] bf16
        const bf16_t* __restrict__ Whh,   // [2][1024][256] bf16
        const float* __restrict__ bih,    // [2][1024]
        const float* __restrict__ bhh,    // [2][1024]
        float* __restrict__ feat)         // [16384][512]
{
  const int dir  = blockIdx.y;
  const int n0   = blockIdx.x * 32;
  const int tid  = threadIdx.x;
  const int wv   = tid >> 5;
  const int lane = tid & 31;
  const int lhalf = lane >> 4;
  const int l16   = lane & 15;

  __shared__ alignas(64) bf16_t sX[2][32 * 768];   // 2 x 48 KB
  __shared__ alignas(32) bf16_t sHb[32 * 256];     // 16 KB

  const bf16_t* Wi = Wih + (size_t)dir * G4 * HH;
  const bf16_t* Wh = Whh + (size_t)dir * G4 * HLc;

  float bias[4][2];
#pragma unroll
  for (int g = 0; g < 4; ++g)
#pragma unroll
    for (int c2 = 0; c2 < 2; ++c2) {
      int col = g * 256 + wv * 32 + c2 * 16 + l16;
      bias[g][c2] = bih[dir * G4 + col] + bhh[dir * G4 + col];
    }

  for (int e = tid; e < 32 * 256; e += 256) sHb[e] = (bf16_t)0.0f;
  v8f creg[2][2];   // [row-tile][col-subtile]
#pragma unroll
  for (int rt = 0; rt < 2; ++rt)
#pragma unroll
    for (int c2 = 0; c2 < 2; ++c2)
#pragma unroll
      for (int j = 0; j < 8; ++j) creg[rt][c2][j] = 0.0f;

  // prefetch first tile into buffer 0 (32*768*2 = 48KB -> 12 chunks/thread)
  {
    const int t0 = dir ? (LL - 1) : 0;
    async_tile_copy(x + ((size_t)t0 * NB + n0) * HH, &sX[0][0], tid, 12);
    async_wait0();
  }
  __syncthreads();

  for (int step = 0; step < LL; ++step) {
    const int buf = step & 1;
    if (step + 1 < LL) {
      const int tn = dir ? (LL - 2 - step) : (step + 1);
      async_tile_copy(x + ((size_t)tn * NB + n0) * HH, &sX[buf ^ 1][0], tid, 12);
    }

    v8f acc[4][2][2];   // [gate][row-tile][col-subtile]
#pragma unroll
    for (int g = 0; g < 4; ++g)
#pragma unroll
      for (int rt = 0; rt < 2; ++rt)
#pragma unroll
        for (int c2 = 0; c2 < 2; ++c2)
#pragma unroll
          for (int j = 0; j < 8; ++j) acc[g][rt][c2][j] = bias[g][c2];

    // input projection: K = 768 (24 x k32) -- keep rolled to avoid spills
#pragma unroll 1
    for (int ki = 0; ki < 24; ++ki) {
      const int k0 = ki * 32;
      v16bf a0 = frag_ld(&sX[buf][l16 * 768 + k0 + lhalf * 16]);
      v16bf a1 = frag_ld(&sX[buf][(16 + l16) * 768 + k0 + lhalf * 16]);
#pragma unroll
      for (int g = 0; g < 4; ++g)
#pragma unroll
        for (int c2 = 0; c2 < 2; ++c2) {
          const int ncol = g * 256 + wv * 32 + c2 * 16 + l16;
          v16bf b = frag_ld(&Wi[(size_t)ncol * HH + k0 + lhalf * 16]);
          acc[g][0][c2] = wmma_bf16(a0, b, acc[g][0][c2]);
          acc[g][1][c2] = wmma_bf16(a1, b, acc[g][1][c2]);
        }
    }
    // recurrent projection: K = 256 (8 x k32), A direct from bf16 h in LDS
#pragma unroll 1
    for (int ki = 0; ki < 8; ++ki) {
      const int k0 = ki * 32;
      v16bf a0 = frag_ld(&sHb[l16 * 256 + k0 + lhalf * 16]);
      v16bf a1 = frag_ld(&sHb[(16 + l16) * 256 + k0 + lhalf * 16]);
#pragma unroll
      for (int g = 0; g < 4; ++g)
#pragma unroll
        for (int c2 = 0; c2 < 2; ++c2) {
          const int ncol = g * 256 + wv * 32 + c2 * 16 + l16;
          v16bf b = frag_ld(&Wh[(size_t)ncol * HLc + k0 + lhalf * 16]);
          acc[g][0][c2] = wmma_bf16(a0, b, acc[g][0][c2]);
          acc[g][1][c2] = wmma_bf16(a1, b, acc[g][1][c2]);
        }
    }
    __syncthreads();   // all waves done reading sHb / sX[buf]

    // gates: torch order i, f, g, o
#pragma unroll
    for (int rt = 0; rt < 2; ++rt)
#pragma unroll
      for (int c2 = 0; c2 < 2; ++c2)
#pragma unroll
        for (int j = 0; j < 8; ++j) {
          float iv = sigm(acc[0][rt][c2][j]);
          float fv = sigm(acc[1][rt][c2][j]);
          float gv = tanhf(acc[2][rt][c2][j]);
          float ov = sigm(acc[3][rt][c2][j]);
          float cv = fv * creg[rt][c2][j] + iv * gv;
          creg[rt][c2][j] = cv;
          float hv = ov * tanhf(cv);
          const int row = rt * 16 + j + lhalf * 8;
          const int col = wv * 32 + c2 * 16 + l16;
          sHb[row * 256 + col] = (bf16_t)hv;
        }
    async_wait0();     // next x tile has landed
    __syncthreads();
  }

  for (int e = tid; e < 32 * 256; e += 256) {
    const int r = e >> 8, c = e & 255;
    feat[(size_t)(n0 + r) * D1c + dir * HLc + c] = (float)sHb[e];
  }
}

// ---------------------------------------------------------------- sent masks
__global__ void k_build_sent(const float* __restrict__ feat,
                             const long long* __restrict__ ttid,
                             const long long* __restrict__ amask,
                             bf16_t* __restrict__ sa, bf16_t* __restrict__ sb) {
  int e = blockIdx.x * blockDim.x + threadIdx.x;
  const int total = TT2 * BB * D1c;
  if (e >= total) return;
  int k = e & 511;
  int b = (e >> 9) & 31;
  int t = e >> 14;                 // [t][b][k]
  float v  = feat[((size_t)b * SS + t + 1) * D1c + k];
  float am = (float)amask[b * SS + t + 1];
  float tt = (float)ttid[b * SS + t + 1];
  sa[e] = (bf16_t)(v * am * tt);
  sb[e] = (bf16_t)(v * am * (1.0f - tt));
}

// ---------------------------------------------------------------- LSTM2 input proj
// xg[sd][t][b][:] = sent @ Wih^T + bih + bhh (bf16 out), sd = sent*2 + dir.
// A tile (32 x 512 bf16) staged by the Tensor Data Mover (TENSORcnt path).
__global__ void __launch_bounds__(256) __attribute__((amdgpu_waves_per_eu(2)))
k_xg_gemm(const bf16_t* __restrict__ sentA, const bf16_t* __restrict__ sentB,
          const bf16_t* __restrict__ Wih,   // [2][1024][512] bf16
          const float* __restrict__ bih, const float* __restrict__ bhh,
          bf16_t* __restrict__ xg)          // [4][511][32][1024] bf16
{
  const int sd   = blockIdx.y;
  const int sent = sd >> 1, dir = sd & 1;
  const int m0   = blockIdx.x * 32;
  const int tid  = threadIdx.x, wv = tid >> 5, lane = tid & 31;
  const int lhalf = lane >> 4, l16 = lane & 15;

  __shared__ alignas(64) bf16_t sA[32 * 512];   // 32 KB

  const bf16_t* src = (sent == 0 ? sentA : sentB) + (size_t)m0 * D1c;
  if (tid < 32) {                               // one wave issues the TDM op
    tdm_load_2d_bf16((unsigned)(size_t)(void*)&sA[0], src, 32, D1c);
    tensor_wait0();
  }
  __syncthreads();

  v8f acc[8][2];    // [col tile][row tile]
  const float* bi = bih + dir * G4;
  const float* bh = bhh + dir * G4;
#pragma unroll
  for (int i = 0; i < 8; ++i) {
    int col = wv * 128 + i * 16 + l16;
    float bv = bi[col] + bh[col];
#pragma unroll
    for (int rt = 0; rt < 2; ++rt)
#pragma unroll
      for (int j = 0; j < 8; ++j) acc[i][rt][j] = bv;
  }

  const bf16_t* W = Wih + (size_t)dir * G4 * D1c;
#pragma unroll 1
  for (int ki = 0; ki < 16; ++ki) {
    int k0 = ki * 32;
    v16bf a0 = frag_ld(&sA[l16 * 512 + k0 + lhalf * 16]);
    v16bf a1 = frag_ld(&sA[(16 + l16) * 512 + k0 + lhalf * 16]);
#pragma unroll
    for (int i = 0; i < 8; ++i) {
      int ncol = wv * 128 + i * 16 + l16;
      v16bf b = frag_ld(&W[(size_t)ncol * D1c + k0 + lhalf * 16]);
      acc[i][0] = wmma_bf16(a0, b, acc[i][0]);
      acc[i][1] = wmma_bf16(a1, b, acc[i][1]);
    }
  }

  bf16_t* out = xg + ((size_t)sd * M2 + m0) * G4;
#pragma unroll
  for (int i = 0; i < 8; ++i) {
    int col = wv * 128 + i * 16 + l16;
#pragma unroll
    for (int rt = 0; rt < 2; ++rt)
#pragma unroll
      for (int j = 0; j < 8; ++j) {
        int row = rt * 16 + j + lhalf * 8;
        out[(size_t)row * G4 + col] = (bf16_t)acc[i][rt][j];
      }
  }
}

// ---------------------------------------------------------------- LSTM2 scan
// 4 blocks, one per (sent, dir). 511 sequential steps; xg already holds the
// input projection + biases; recurrence via WMMA each step.
__global__ void __launch_bounds__(256) __attribute__((amdgpu_waves_per_eu(2)))
k_lstm2(const bf16_t* __restrict__ xg,    // [4][511][32][1024] bf16
        const bf16_t* __restrict__ Whh,   // [2][1024][256] bf16
        float* __restrict__ emb)          // [4][32][256]
{
  const int sd  = blockIdx.x;
  const int dir = sd & 1;
  const int tid = threadIdx.x, wv = tid >> 5, lane = tid & 31;
  const int lhalf = lane >> 4, l16 = lane & 15;

  __shared__ alignas(32) bf16_t sHb[32 * 256];  // 16 KB

  const bf16_t* Wh  = Whh + (size_t)dir * G4 * HLc;
  const bf16_t* xgb = xg + (size_t)sd * M2 * G4;

  for (int e = tid; e < 32 * 256; e += 256) sHb[e] = (bf16_t)0.0f;
  v8f creg[2][2];
#pragma unroll
  for (int rt = 0; rt < 2; ++rt)
#pragma unroll
    for (int c2 = 0; c2 < 2; ++c2)
#pragma unroll
      for (int j = 0; j < 8; ++j) creg[rt][c2][j] = 0.0f;
  __syncthreads();

#pragma unroll 1
  for (int step = 0; step < TT2; ++step) {
    const int t = dir ? (TT2 - 1 - step) : step;
    const bf16_t* xt = xgb + (size_t)t * BB * G4;

    v8f acc[4][2][2];   // [gate][row-tile][col-subtile]
#pragma unroll
    for (int g = 0; g < 4; ++g)
#pragma unroll
      for (int rt = 0; rt < 2; ++rt)
#pragma unroll
        for (int c2 = 0; c2 < 2; ++c2) {
          int col = g * 256 + wv * 32 + c2 * 16 + l16;
#pragma unroll
          for (int j = 0; j < 8; ++j) {
            int row = rt * 16 + j + lhalf * 8;
            acc[g][rt][c2][j] = (float)xt[(size_t)row * G4 + col];
          }
        }

#pragma unroll 1
    for (int ki = 0; ki < 8; ++ki) {
      int k0 = ki * 32;
      v16bf a0 = frag_ld(&sHb[l16 * 256 + k0 + lhalf * 16]);
      v16bf a1 = frag_ld(&sHb[(16 + l16) * 256 + k0 + lhalf * 16]);
#pragma unroll
      for (int g = 0; g < 4; ++g)
#pragma unroll
        for (int c2 = 0; c2 < 2; ++c2) {
          int ncol = g * 256 + wv * 32 + c2 * 16 + l16;
          v16bf b = frag_ld(&Wh[(size_t)ncol * HLc + k0 + lhalf * 16]);
          acc[g][0][c2] = wmma_bf16(a0, b, acc[g][0][c2]);
          acc[g][1][c2] = wmma_bf16(a1, b, acc[g][1][c2]);
        }
    }
    __syncthreads();   // done reading old sHb

#pragma unroll
    for (int rt = 0; rt < 2; ++rt)
#pragma unroll
      for (int c2 = 0; c2 < 2; ++c2)
#pragma unroll
        for (int j = 0; j < 8; ++j) {
          float iv = sigm(acc[0][rt][c2][j]);
          float fv = sigm(acc[1][rt][c2][j]);
          float gv = tanhf(acc[2][rt][c2][j]);
          float ov = sigm(acc[3][rt][c2][j]);
          float cv = fv * creg[rt][c2][j] + iv * gv;
          creg[rt][c2][j] = cv;
          float hv = ov * tanhf(cv);
          int row = rt * 16 + j + lhalf * 8;
          int col = wv * 32 + c2 * 16 + l16;
          sHb[row * 256 + col] = (bf16_t)hv;
        }
    __syncthreads();
  }

  for (int e = tid; e < 32 * 256; e += 256)
    emb[(size_t)sd * 32 * 256 + e] = (float)sHb[e];
}

// ---------------------------------------------------------------- head
__global__ void __launch_bounds__(256)
k_head(const float* __restrict__ feat, const float* __restrict__ emb,
       const float* __restrict__ W1, const float* __restrict__ b1,
       const float* __restrict__ bng, const float* __restrict__ bnb,
       const float* __restrict__ W2, const float* __restrict__ b2,
       float* __restrict__ out)
{
  __shared__ float sH1[BB][256];
  const int j = threadIdx.x;                 // output column 0..255
  float h[BB];
  const float* w = W1 + (size_t)j * 1536;
  for (int b = 0; b < BB; ++b) {
    float s = b1[j];
    for (int k = 0; k < 512; ++k)            // cls_emb = lw_out[0][b]
      s += feat[((size_t)b * SS) * D1c + k] * w[k];
    for (int q = 0; q < 1024; ++q) {         // emb_a (sd 0,1) then emb_b (sd 2,3)
      int sd = q >> 8, c = q & 255;
      s += emb[((size_t)sd * BB + b) * 256 + c] * w[512 + q];
    }
    h[b] = s;
  }
  float mu = 0.f;
  for (int b = 0; b < BB; ++b) mu += h[b];
  mu *= (1.0f / BB);
  float var = 0.f;
  for (int b = 0; b < BB; ++b) { float d = h[b] - mu; var += d * d; }
  var *= (1.0f / BB);
  float inv = rsqrtf(var + 1e-5f);
  for (int b = 0; b < BB; ++b) {
    float v = (h[b] - mu) * inv * bng[j] + bnb[j];
    sH1[b][j] = (v >= 0.f) ? v : 0.01f * v;
  }
  __syncthreads();
  if (j < BB * 2) {
    int b = j >> 1, o = j & 1;
    float s = b2[o];
    for (int k = 0; k < 256; ++k) s += sH1[b][k] * W2[o * 256 + k];
    out[b * 2 + o] = s;
  }
}

// ---------------------------------------------------------------- launch
extern "C" void kernel_launch(void* const* d_in, const int* in_sizes, int n_in,
                              void* d_out, int out_size, void* d_ws, size_t ws_size,
                              hipStream_t stream) {
  (void)in_sizes; (void)n_in; (void)out_size; (void)ws_size;
  const float*     hs    = (const float*)d_in[0];
  const long long* ttid  = (const long long*)d_in[1];
  const long long* amask = (const long long*)d_in[2];
  const float* lwWih = (const float*)d_in[3];
  const float* lwWhh = (const float*)d_in[4];
  const float* lwBih = (const float*)d_in[5];
  const float* lwBhh = (const float*)d_in[6];
  const float* twWih = (const float*)d_in[7];
  const float* twWhh = (const float*)d_in[8];
  const float* twBih = (const float*)d_in[9];
  const float* twBhh = (const float*)d_in[10];
  const float* W1  = (const float*)d_in[11];
  const float* b1  = (const float*)d_in[12];
  const float* bng = (const float*)d_in[13];
  const float* bnb = (const float*)d_in[14];
  const float* W2  = (const float*)d_in[15];
  const float* b2  = (const float*)d_in[16];
  float* out = (float*)d_out;

  char* ws = (char*)d_ws;
  size_t off = 0;
  auto alloc = [&](size_t bytes) {
    size_t o = off; off += (bytes + 255) & ~(size_t)255; return o;
  };
  bf16_t* lwWih_h = (bf16_t*)(ws + alloc((size_t)2 * G4 * HH  * 2));
  bf16_t* lwWhh_h = (bf16_t*)(ws + alloc((size_t)2 * G4 * HLc * 2));
  bf16_t* twWih_h = (bf16_t*)(ws + alloc((size_t)2 * G4 * D1c * 2));
  bf16_t* twWhh_h = (bf16_t*)(ws + alloc((size_t)2 * G4 * HLc * 2));
  bf16_t* x_h     = (bf16_t*)(ws + alloc((size_t)LL * NB * HH * 2));
  float*  feat    = (float*) (ws + alloc((size_t)NB * D1c * 4));
  bf16_t* sentA   = (bf16_t*)(ws + alloc((size_t)M2 * D1c * 2));
  bf16_t* sentB   = (bf16_t*)(ws + alloc((size_t)M2 * D1c * 2));
  bf16_t* xg2     = (bf16_t*)(ws + alloc((size_t)4 * M2 * G4 * 2));
  float*  emb     = (float*) (ws + alloc((size_t)4 * BB * HLc * 4));

  k_cvt<<<(2 * G4 * HH  + 255) / 256, 256, 0, stream>>>(lwWih, lwWih_h, 2 * G4 * HH);
  k_cvt<<<(2 * G4 * HLc + 255) / 256, 256, 0, stream>>>(lwWhh, lwWhh_h, 2 * G4 * HLc);
  k_cvt<<<(2 * G4 * D1c + 255) / 256, 256, 0, stream>>>(twWih, twWih_h, 2 * G4 * D1c);
  k_cvt<<<(2 * G4 * HLc + 255) / 256, 256, 0, stream>>>(twWhh, twWhh_h, 2 * G4 * HLc);
  {
    const int nx = LL * NB * HH;  // 163,577,856
    k_cvt<<<(nx + 255) / 256, 256, 0, stream>>>(hs, x_h, nx);
  }

  k_lstm1<<<dim3(NB / 32, 2), 256, 0, stream>>>(x_h, lwWih_h, lwWhh_h,
                                                lwBih, lwBhh, feat);

  k_build_sent<<<(TT2 * BB * D1c + 255) / 256, 256, 0, stream>>>(feat, ttid, amask,
                                                                 sentA, sentB);

  k_xg_gemm<<<dim3(M2 / 32, 4), 256, 0, stream>>>(sentA, sentB, twWih_h,
                                                  twBih, twBhh, xg2);

  k_lstm2<<<4, 256, 0, stream>>>(xg2, twWhh_h, emb);

  k_head<<<1, 256, 0, stream>>>(feat, emb, W1, b1, bng, bnb, W2, b2, out);
}
